// SMSA_87196426043490
// MI455X (gfx1250) — compile-verified, matrix operand order checked by reading, not run on hardware
//
#include <hip/hip_runtime.h>
#include <hip/hip_bf16.h>

typedef __attribute__((ext_vector_type(16))) _Float16 v16h;
typedef __attribute__((ext_vector_type(8)))  float    v8f;
typedef __attribute__((ext_vector_type(4)))  unsigned int u32x4;
typedef __attribute__((ext_vector_type(8)))  int      i32x8;
typedef __attribute__((ext_vector_type(4)))  int      i32x4;

#if __has_builtin(__builtin_amdgcn_tensor_load_to_lds)
#define HAVE_TDM 1
#else
#define HAVE_TDM 0
#endif

// ---------------------------------------------------------------------------
// WMMA helpers (gfx1250, wave32).
// A-fragment (16x32 f16): lane L holds row M=L&15; K pattern:
//   elements 0..7  -> K = kb+0..7,   elements 8..15 -> K = kb+16..23,
//   kb = 0 for lanes 0..15, kb = 8 for lanes 16..31.
// B (32x16) is loaded column-major: same pattern with "major" = column N.
// C/D (16x16 f32): lane L holds col N=L&15; element r holds row M=r+(L>=16?8:0).
// ---------------------------------------------------------------------------
static __device__ inline v16h ld_frag(const _Float16* tile, int majStride, int kStride) {
  const int lane = threadIdx.x & 31;
  const _Float16* p = tile + (lane & 15) * majStride + ((lane >> 4) ? 8 * kStride : 0);
  v16h f;
#pragma unroll
  for (int i = 0; i < 8; ++i) f[i] = p[i * kStride];
#pragma unroll
  for (int i = 0; i < 8; ++i) f[8 + i] = p[(16 + i) * kStride];
  return f;
}

static __device__ inline v8f wmma16(v16h a, v16h b, v8f c) {
  return __builtin_amdgcn_wmma_f32_16x16x32_f16(false, a, false, b, (short)0, c, false, false);
}

static __device__ inline float fsqrt(float x) { return __builtin_amdgcn_sqrtf(x); }

// ---------------------------------------------------------------------------
// TDM: contiguous 1D copy global->LDS, data_size=8B. n8 = number of 8B units.
// Descriptor per CDNA5 ISA 8.3/8.4: group0 {count=1, lds_addr, global_addr,
// type=2}; group1 {data_size=3, tensor_dim0=n8, tensor_dim1=1, tile_dim0=n8,
// stride0=n8}; groups 2/3 zero (<=2D).
// ---------------------------------------------------------------------------
#if HAVE_TDM
static __device__ inline void tdm_copy_1d(unsigned int lds_off, const void* gsrc,
                                          unsigned int n8) {
  unsigned long long ga = (unsigned long long)(uintptr_t)gsrc;
  u32x4 g0;
  g0[0] = 1u;                                                   // count=1
  g0[1] = lds_off;                                              // lds byte addr
  g0[2] = (unsigned int)ga;                                     // gaddr[31:0]
  g0[3] = (unsigned int)((ga >> 32) & 0x1ffffffu) | (2u << 30); // gaddr[56:32]|type=2
  i32x8 g1;
  g1[0] = (int)(3u << 16);                       // data_size=3 (8B units)
  g1[1] = (int)((n8 & 0xffffu) << 16);           // tensor_dim0[15:0]
  g1[2] = (int)(((n8 >> 16) & 0xffffu) | (1u << 16)); // tensor_dim0[31:16] | tensor_dim1=1
  g1[3] = (int)((n8 & 0xffffu) << 16);           // tile_dim0 = n8
  g1[4] = 0;                                     // tile_dim1=0, tile_dim2=0
  g1[5] = (int)n8;                               // tensor_dim0_stride[31:0]
  g1[6] = 0;
  g1[7] = 0;
  i32x4 z4 = {0, 0, 0, 0};
#if __clang_major__ >= 23
  i32x8 z8 = {0, 0, 0, 0, 0, 0, 0, 0};
  __builtin_amdgcn_tensor_load_to_lds(g0, g1, z4, z4, z8, 0);
#else
  __builtin_amdgcn_tensor_load_to_lds(g0, g1, z4, z4, 0);
#endif
}
#endif

// ---------------------------------------------------------------------------
// Weight normalization: W[r,:] = v[r,:] * g[r] / ||v[r,:]|| , emitted as f16.
// ---------------------------------------------------------------------------
__global__ void prep_weights(const float* vq, const float* gq,
                             const float* v1, const float* g1,
                             const float* v2, const float* g2,
                             _Float16* Wq, _Float16* W1, _Float16* W2) {
  __shared__ float sm[4];
  int r = blockIdx.x;
  const float* v; float g; _Float16* dst;
  if (r < 384)       { v = vq + (size_t)r * 128;        g = gq[r];       dst = Wq + (size_t)r * 128; }
  else if (r < 512)  { int rr = r - 384; v = v1 + (size_t)rr * 128; g = g1[rr]; dst = W1 + (size_t)rr * 128; }
  else               { int rr = r - 512; v = v2 + (size_t)rr * 128; g = g2[rr]; dst = W2 + (size_t)rr * 128; }
  float e = v[threadIdx.x];
  float s = e * e;
#pragma unroll
  for (int m = 16; m >= 1; m >>= 1) s += __shfl_xor(s, m, 32);
  if ((threadIdx.x & 31) == 0) sm[threadIdx.x >> 5] = s;
  __syncthreads();
  float nrm = sqrtf(sm[0] + sm[1] + sm[2] + sm[3]);
  dst[threadIdx.x] = (_Float16)(e * g / nrm);
}

// x fp32 -> f16 copy + per-row squared norms. One block per row (8192 rows).
__global__ void prep_x(const float* x, _Float16* xh, float* sq) {
  __shared__ float sm[4];
  size_t row = blockIdx.x;
  float e = x[row * 128 + threadIdx.x];
  xh[row * 128 + threadIdx.x] = (_Float16)e;
  float s = e * e;
#pragma unroll
  for (int m = 16; m >= 1; m >>= 1) s += __shfl_xor(s, m, 32);
  if ((threadIdx.x & 31) == 0) sm[threadIdx.x >> 5] = s;
  __syncthreads();
  if (threadIdx.x == 0) sq[row] = sm[0] + sm[1] + sm[2] + sm[3];
}

// Spatial sigma: per grid-row partial sums of pairwise grid distances.
__global__ void spa_partial(float* part) {
  __shared__ float sm[8];
  int i = blockIdx.x;
  float xi = (float)(i >> 6), yi = (float)(i & 63);
  float s = 0.f;
  for (int j = threadIdx.x; j < 4096; j += 256) {
    float dx = xi - (float)(j >> 6), dy = yi - (float)(j & 63);
    s += fsqrt(dx * dx + dy * dy);
  }
#pragma unroll
  for (int m = 16; m >= 1; m >>= 1) s += __shfl_xor(s, m, 32);
  if ((threadIdx.x & 31) == 0) sm[threadIdx.x >> 5] = s;
  __syncthreads();
  if (threadIdx.x == 0) {
    float t = 0.f;
#pragma unroll
    for (int w = 0; w < 8; ++w) t += sm[w];
    part[i] = t;
  }
}

// Spectral sigma pass 1: per 64x64 tile, G = x.x^T via WMMA, sum sqrt(max(d2,0)).
__global__ void __launch_bounds__(128) spe_partial(const _Float16* xh, const float* sq, float* part) {
  __shared__ float sm[4];
  int it = blockIdx.x, jt = blockIdx.y, b = blockIdx.z;
  int w = threadIdx.x >> 5, lane = threadIdx.x & 31;
  int half = lane >> 4, n16 = lane & 15;
  int i0 = it * 64 + w * 16;
  const _Float16* xb = xh + (size_t)b * 4096 * 128;
  v16h xa[4];
#pragma unroll
  for (int kc = 0; kc < 4; ++kc) xa[kc] = ld_frag(xb + (size_t)i0 * 128 + kc * 32, 128, 1);
  float si[8];
#pragma unroll
  for (int r = 0; r < 8; ++r) si[r] = sq[b * 4096 + i0 + r + half * 8];
  float acc = 0.f;
  for (int q4 = 0; q4 < 4; ++q4) {
    int j0 = jt * 64 + q4 * 16;
    v16h b0 = ld_frag(xb + (size_t)j0 * 128 + 0 * 32, 128, 1);
    v16h b1 = ld_frag(xb + (size_t)j0 * 128 + 1 * 32, 128, 1);
    v16h b2 = ld_frag(xb + (size_t)j0 * 128 + 2 * 32, 128, 1);
    v16h b3 = ld_frag(xb + (size_t)j0 * 128 + 3 * 32, 128, 1);
    v8f g = {};
    g = wmma16(xa[0], b0, g);
    g = wmma16(xa[1], b1, g);
    g = wmma16(xa[2], b2, g);
    g = wmma16(xa[3], b3, g);
    float sqj = sq[b * 4096 + j0 + n16];
#pragma unroll
    for (int r = 0; r < 8; ++r) {
      float d2 = si[r] + sqj - 2.f * g[r];
      acc += fsqrt(fmaxf(d2, 0.f));
    }
  }
#pragma unroll
  for (int m = 16; m >= 1; m >>= 1) acc += __shfl_xor(acc, m, 32);
  if (lane == 0) sm[w] = acc;
  __syncthreads();
  if (threadIdx.x == 0) part[((b * 64 + it) * 64) + jt] = sm[0] + sm[1] + sm[2] + sm[3];
}

// Deterministic final reduction -> consts = { 1/(2*sig_spe^2), 1/(2*sig_spa^2) }.
__global__ void finalize_sigma(const float* pspe, const float* pspa, float* consts) {
  __shared__ float sm[8];
  __shared__ float tot1s;
  float s1 = 0.f;
  for (int i = threadIdx.x; i < 8192; i += 256) s1 += pspe[i];
#pragma unroll
  for (int m = 16; m >= 1; m >>= 1) s1 += __shfl_xor(s1, m, 32);
  if ((threadIdx.x & 31) == 0) sm[threadIdx.x >> 5] = s1;
  __syncthreads();
  if (threadIdx.x == 0) {
    float t = 0.f;
#pragma unroll
    for (int w = 0; w < 8; ++w) t += sm[w];
    tot1s = t;
  }
  __syncthreads();
  float s2 = 0.f;
  for (int i = threadIdx.x; i < 4096; i += 256) s2 += pspa[i];
#pragma unroll
  for (int m = 16; m >= 1; m >>= 1) s2 += __shfl_xor(s2, m, 32);
  if ((threadIdx.x & 31) == 0) sm[threadIdx.x >> 5] = s2;
  __syncthreads();
  if (threadIdx.x == 0) {
    float tot2 = 0.f;
#pragma unroll
    for (int w = 0; w < 8; ++w) tot2 += sm[w];
    float sg1 = tot1s / 33554432.f;   // b*n*n
    float sg2 = tot2  / 16777216.f;   // n*n
    consts[0] = 1.f / (2.f * sg1 * sg1);
    consts[1] = 1.f / (2.f * sg2 * sg2);
  }
}

// ---------------------------------------------------------------------------
// Generic weight-normed GEMM:  Y[M,N] = A[M,128] (f16) * W[N,128]^T + bias.
// mode 0: scatter to q/k/v [b,h,n,64] f16.  mode 1: GELU(exact) -> f16.
// mode 2: fp32 -> d_out.
// ---------------------------------------------------------------------------
__global__ void __launch_bounds__(128) gemm_wn(const _Float16* A, const _Float16* W,
                                               const float* bias, int mode,
                                               _Float16* q, _Float16* k, _Float16* v,
                                               _Float16* outH, float* outF) {
  int w = threadIdx.x >> 5, lane = threadIdx.x & 31;
  int half = lane >> 4, n16 = lane & 15;
  int row0 = blockIdx.x * 64 + w * 16;
  int nb0 = blockIdx.y * 64;
  v16h a[4];
#pragma unroll
  for (int kc = 0; kc < 4; ++kc) a[kc] = ld_frag(A + (size_t)row0 * 128 + kc * 32, 128, 1);
  v8f acc[4] = {};
#pragma unroll
  for (int kc = 0; kc < 4; ++kc) {
    v16h b0 = ld_frag(W + (size_t)(nb0 + 0)  * 128 + kc * 32, 128, 1);
    v16h b1 = ld_frag(W + (size_t)(nb0 + 16) * 128 + kc * 32, 128, 1);
    v16h b2 = ld_frag(W + (size_t)(nb0 + 32) * 128 + kc * 32, 128, 1);
    v16h b3 = ld_frag(W + (size_t)(nb0 + 48) * 128 + kc * 32, 128, 1);
    acc[0] = wmma16(a[kc], b0, acc[0]);
    acc[1] = wmma16(a[kc], b1, acc[1]);
    acc[2] = wmma16(a[kc], b2, acc[2]);
    acc[3] = wmma16(a[kc], b3, acc[3]);
  }
#pragma unroll
  for (int nt = 0; nt < 4; ++nt) {
    int col = nb0 + nt * 16 + n16;
    float bv = bias[col];
#pragma unroll
    for (int r = 0; r < 8; ++r) {
      int gm = row0 + r + half * 8;
      float val = acc[nt][r] + bv;
      if (mode == 0) {
        int bb = gm >> 12, nn = gm & 4095;
        int sect = col >> 7;           // 0:q 1:k 2:v
        int hh = (col >> 6) & 1;
        int dd = col & 63;
        _Float16* dst = (sect == 0) ? q : ((sect == 1) ? k : v);
        dst[(((size_t)(bb * 2 + hh)) * 4096 + nn) * 64 + dd] = (_Float16)val;
      } else if (mode == 1) {
        float ge = 0.5f * val * (1.f + erff(val * 0.70710678118654752f));
        outH[(size_t)gm * 128 + col] = (_Float16)ge;
      } else {
        outF[(size_t)gm * 128 + col] = val;
      }
    }
  }
}

// ---------------------------------------------------------------------------
// Fused masked flash attention. grid (n/64, h, b), 128 threads (4 waves).
// TDM double-buffers K/V/Xj tiles (wave 0 issues, s_wait_tensorcnt pipelines),
// S = q.k^T (WMMA), G = x_i.x_j^T (WMMA), masks + online softmax, P.V (WMMA).
// LDS map (single block so TDM lds_addr == offset 0):
//   [0,16K)   buffer 0 : K(4K) V(4K) X(8K)
//   [16K,32K) buffer 1 : K(4K) V(4K) X(8K)
//   [32K,36K) P staging (4 waves x 16x32 f16)
// ---------------------------------------------------------------------------
__global__ void __launch_bounds__(128) flash_attn(const _Float16* qb, const _Float16* kb,
                                                  const _Float16* vb, const _Float16* xh,
                                                  const float* sq, const float* consts,
                                                  _Float16* attnH) {
  __shared__ alignas(128) unsigned char smem[2 * 16384 + 4096];

  int w = threadIdx.x >> 5, lane = threadIdx.x & 31;
  int half = lane >> 4, n16 = lane & 15;
  int h = blockIdx.y, b = blockIdx.z;
  int i0 = blockIdx.x * 64 + w * 16;
  const float cspe = consts[0], cspa = consts[1];
  const float scale = 0.125f;   // 64^-0.5

  const _Float16* qrow = qb + (((size_t)(b * 2 + h)) * 4096 + i0) * 64;
  const _Float16* krow = kb + ((size_t)(b * 2 + h)) * 4096 * 64;
  const _Float16* vrow = vb + ((size_t)(b * 2 + h)) * 4096 * 64;
  const _Float16* xrow = xh + (size_t)b * 4096 * 128;
  const float*    sqb  = sq + b * 4096;
  _Float16* Pst = (_Float16*)(smem + 32768);

  v16h qa[2], xa[4];
#pragma unroll
  for (int kc = 0; kc < 2; ++kc) qa[kc] = ld_frag(qrow + kc * 32, 64, 1);
#pragma unroll
  for (int kc = 0; kc < 4; ++kc) xa[kc] = ld_frag(xrow + (size_t)i0 * 128 + kc * 32, 128, 1);

  float si[8], gxi[8], gyi[8], mrow[8], lrow[8];
  v8f O[4] = {};
#pragma unroll
  for (int r = 0; r < 8; ++r) {
    int gi = i0 + r + half * 8;
    si[r] = sqb[gi];
    gxi[r] = (float)(gi >> 6); gyi[r] = (float)(gi & 63);
    mrow[r] = -1e30f; lrow[r] = 0.f;
  }

#if HAVE_TDM
  // prime buffer 0 with tile j=0 (wave 0 issues; TENSORcnt tracks in-order)
  if (threadIdx.x < 32) {
    tdm_copy_1d(0u,         krow, 512);            // K tile 32x64 f16 = 4KB
    tdm_copy_1d(4096u,      vrow, 512);            // V tile
    tdm_copy_1d(8192u,      xrow, 1024);           // Xj tile 32x128 f16 = 8KB
  }
#endif

  int cur = 0;
  for (int jt = 0; jt < 128; ++jt) {
    int j = jt * 32;
#if HAVE_TDM
    if (threadIdx.x < 32) {
      if (jt + 1 < 128) {
        unsigned int bo = (unsigned int)((cur ^ 1) * 16384);
        tdm_copy_1d(bo,         krow + (size_t)(j + 32) * 64, 512);
        tdm_copy_1d(bo + 4096u, vrow + (size_t)(j + 32) * 64, 512);
        tdm_copy_1d(bo + 8192u, xrow + (size_t)(j + 32) * 128, 1024);
        __builtin_amdgcn_s_wait_tensorcnt((short)3);   // current 3 done, next 3 in flight
      } else {
        __builtin_amdgcn_s_wait_tensorcnt((short)0);
      }
    }
    __syncthreads();
#else
    __syncthreads();
    { // synchronous cooperative loads into current buffer
      const uint32_t* Kg = (const uint32_t*)(krow + (size_t)j * 64);
      const uint32_t* Vg = (const uint32_t*)(vrow + (size_t)j * 64);
      const uint32_t* Xg = (const uint32_t*)(xrow + (size_t)j * 128);
      uint32_t* Ks = (uint32_t*)(smem + cur * 16384);
      uint32_t* Vs = (uint32_t*)(smem + cur * 16384 + 4096);
      uint32_t* Xs = (uint32_t*)(smem + cur * 16384 + 8192);
#pragma unroll
      for (int t = 0; t < 8; ++t)  Ks[threadIdx.x + t * 128] = Kg[threadIdx.x + t * 128];
#pragma unroll
      for (int t = 0; t < 8; ++t)  Vs[threadIdx.x + t * 128] = Vg[threadIdx.x + t * 128];
#pragma unroll
      for (int t = 0; t < 16; ++t) Xs[threadIdx.x + t * 128] = Xg[threadIdx.x + t * 128];
    }
    __syncthreads();
#endif
    const _Float16* Kt = (const _Float16*)(smem + cur * 16384);
    const _Float16* Vt = (const _Float16*)(smem + cur * 16384 + 4096);
    const _Float16* Xj = (const _Float16*)(smem + cur * 16384 + 8192);
    if (jt + 2 < 128) __builtin_prefetch(krow + (size_t)(j + 64) * 64, 0, 1);

    v8f st[2];
#pragma unroll
    for (int t = 0; t < 2; ++t) {
      v16h kf0 = ld_frag(Kt + (t * 16) * 64, 64, 1);
      v16h kf1 = ld_frag(Kt + (t * 16) * 64 + 32, 64, 1);
      v16h xf0 = ld_frag(Xj + (t * 16) * 128 + 0 * 32, 128, 1);
      v16h xf1 = ld_frag(Xj + (t * 16) * 128 + 1 * 32, 128, 1);
      v16h xf2 = ld_frag(Xj + (t * 16) * 128 + 2 * 32, 128, 1);
      v16h xf3 = ld_frag(Xj + (t * 16) * 128 + 3 * 32, 128, 1);
      v8f s = {};
      s = wmma16(qa[0], kf0, s);
      s = wmma16(qa[1], kf1, s);
      v8f g = {};
      g = wmma16(xa[0], xf0, g);
      g = wmma16(xa[1], xf1, g);
      g = wmma16(xa[2], xf2, g);
      g = wmma16(xa[3], xf3, g);
      int jc = j + t * 16 + n16;
      float sqjv = sqb[jc];
      float gxj = (float)(jc >> 6), gyj = (float)(jc & 63);
#pragma unroll
      for (int r = 0; r < 8; ++r) {
        float d2 = si[r] + sqjv - 2.f * g[r];
        float spe = __expf(-fsqrt(fmaxf(d2, 0.f)) * cspe);
        float dx = gxi[r] - gxj, dy = gyi[r] - gyj;
        float spa = __expf(-fsqrt(dx * dx + dy * dy) * cspa);
        s[r] = s[r] * scale * spe * spa;
      }
      st[t] = s;
    }

    // online softmax (row state lives in 8 C-layout rows per lane)
    float pr0[8], pr1[8];
#pragma unroll
    for (int r = 0; r < 8; ++r) {
      float cm = fmaxf(st[0][r], st[1][r]);
#pragma unroll
      for (int m = 1; m < 16; m <<= 1) cm = fmaxf(cm, __shfl_xor(cm, m, 32));
      float mn = fmaxf(mrow[r], cm);
      float alpha = __expf(mrow[r] - mn);
      float p0 = __expf(st[0][r] - mn);
      float p1 = __expf(st[1][r] - mn);
      float ls = p0 + p1;
#pragma unroll
      for (int m = 1; m < 16; m <<= 1) ls += __shfl_xor(ls, m, 32);
      lrow[r] = lrow[r] * alpha + ls;
      mrow[r] = mn;
#pragma unroll
      for (int tt = 0; tt < 4; ++tt) O[tt][r] *= alpha;
      pr0[r] = p0; pr1[r] = p1;
    }

    // stage P (C-layout -> A-layout) through per-wave LDS
    _Float16* Pw = Pst + w * 16 * 32;
#pragma unroll
    for (int r = 0; r < 8; ++r) {
      int m = r + half * 8;
      Pw[m * 32 + n16]      = (_Float16)pr0[r];
      Pw[m * 32 + 16 + n16] = (_Float16)pr1[r];
    }
    asm volatile("s_wait_dscnt 0" ::: "memory");   // wave-synchronous LDS RAW
    v16h pf = ld_frag(Pw, 32, 1);
    {
      v16h vf0 = ld_frag(Vt + 0,  1, 64);
      v16h vf1 = ld_frag(Vt + 16, 1, 64);
      v16h vf2 = ld_frag(Vt + 32, 1, 64);
      v16h vf3 = ld_frag(Vt + 48, 1, 64);
      O[0] = wmma16(pf, vf0, O[0]);
      O[1] = wmma16(pf, vf1, O[1]);
      O[2] = wmma16(pf, vf2, O[2]);
      O[3] = wmma16(pf, vf3, O[3]);
    }
    __syncthreads();   // all waves done with `cur` before its TDM overwrite
    cur ^= 1;
  }

  // epilogue: O / l -> attention output f16 [b*n, 128] (heads concatenated)
#pragma unroll
  for (int r = 0; r < 8; ++r) {
    float inv = __builtin_amdgcn_rcpf(lrow[r]);
    int gi = i0 + r + half * 8;
#pragma unroll
    for (int tt = 0; tt < 4; ++tt) {
      float o = O[tt][r] * inv;
      attnH[((size_t)(b * 4096 + gi)) * 128 + h * 64 + tt * 16 + n16] = (_Float16)o;
    }
  }
}

// ---------------------------------------------------------------------------
extern "C" void kernel_launch(void* const* d_in, const int* in_sizes, int n_in,
                              void* d_out, int out_size, void* d_ws, size_t ws_size,
                              hipStream_t stream) {
  (void)in_sizes; (void)n_in; (void)out_size; (void)ws_size;
  const float* x    = (const float*)d_in[0];
  const float* vqkv = (const float*)d_in[1];
  const float* gqkv = (const float*)d_in[2];
  const float* bqkv = (const float*)d_in[3];
  const float* vff1 = (const float*)d_in[4];
  const float* gff1 = (const float*)d_in[5];
  const float* bff1 = (const float*)d_in[6];
  const float* vff2 = (const float*)d_in[7];
  const float* gff2 = (const float*)d_in[8];
  const float* bff2 = (const float*)d_in[9];

  char* ws = (char*)d_ws;
  size_t off = 0;
  auto carve = [&](size_t bytes) -> void* {
    void* p = ws + off;
    off = (off + bytes + 255) & ~(size_t)255;
    return p;
  };
  _Float16* xh    = (_Float16*)carve(8192 * 128 * 2);
  float*    sq    = (float*)   carve(8192 * 4);
  _Float16* Wq    = (_Float16*)carve(384 * 128 * 2);
  _Float16* W1    = (_Float16*)carve(128 * 128 * 2);
  _Float16* W2    = (_Float16*)carve(128 * 128 * 2);
  _Float16* qbuf  = (_Float16*)carve((size_t)2 * 2 * 4096 * 64 * 2);
  _Float16* kbuf  = (_Float16*)carve((size_t)2 * 2 * 4096 * 64 * 2);
  _Float16* vbuf  = (_Float16*)carve((size_t)2 * 2 * 4096 * 64 * 2);
  _Float16* attnH = (_Float16*)carve(8192 * 128 * 2);
  _Float16* h1    = (_Float16*)carve(8192 * 128 * 2);
  float*    pspe  = (float*)   carve(8192 * 4);
  float*    pspa  = (float*)   carve(4096 * 4);
  float*    consts= (float*)   carve(256);

  prep_weights<<<640, 128, 0, stream>>>(vqkv, gqkv, vff1, gff1, vff2, gff2, Wq, W1, W2);
  prep_x<<<8192, 128, 0, stream>>>(x, xh, sq);
  spa_partial<<<4096, 256, 0, stream>>>(pspa);
  spe_partial<<<dim3(64, 64, 2), 128, 0, stream>>>(xh, sq, pspe);
  finalize_sigma<<<1, 256, 0, stream>>>(pspe, pspa, consts);
  gemm_wn<<<dim3(128, 6), 128, 0, stream>>>(xh, Wq, bqkv, 0, qbuf, kbuf, vbuf, nullptr, nullptr);
  flash_attn<<<dim3(64, 2, 2), 128, 0, stream>>>(qbuf, kbuf, vbuf, xh, sq, consts, attnH);
  gemm_wn<<<dim3(128, 2), 128, 0, stream>>>(attnH, W1, bff1, 1, nullptr, nullptr, nullptr, h1, nullptr);
  gemm_wn<<<dim3(128, 2), 128, 0, stream>>>(h1, W2, bff2, 2, nullptr, nullptr, nullptr, nullptr, (float*)d_out);
}
